// SPELL_bak_53953379173229
// MI455X (gfx1250) — matrix operand order repaired and checked by reading, not compiled, not run on hardware
//
#include <hip/hip_runtime.h>

#define N_NODES 30000
#define N_EDGES 960000
#define IN_DIM  1792
#define C0      64
#define C1      64

typedef _Float16 half8 __attribute__((ext_vector_type(8)));
typedef _Float16 v16h  __attribute__((ext_vector_type(16)));
typedef float    v8f   __attribute__((ext_vector_type(8)));
typedef float    v2f   __attribute__((ext_vector_type(2)));

#if defined(__gfx1250__) && __has_builtin(__builtin_amdgcn_wmma_f32_16x16x4_f32)
#define HAVE_WMMA_F32X4 1
#else
#define HAVE_WMMA_F32X4 0
#endif

// ---- monotone float <-> uint encoding for atomic max over signed floats ----
__device__ __forceinline__ unsigned enc_f32(float f) {
  unsigned u = __float_as_uint(f);
  return (u & 0x80000000u) ? ~u : (u | 0x80000000u);
}
#define ENC_NEG_INF 0x007FFFFFu  // enc(-inf)

__device__ __forceinline__ bool edge_mask(float a) {
  return (a == 111.0f) || (fabsf(a) <= 0.05f);
}

// ---------------------------------------------------------------- init ws ---
__global__ __launch_bounds__(256) void k_init(float* nsum, float* cnt, unsigned* agg) {
  int t = blockIdx.x * 256 + threadIdx.x;
  if (t < N_NODES * C0) nsum[t] = 0.0f;
  if (t < N_NODES)      cnt[t]  = 0.0f;
  if (t < N_NODES * 2)  agg[t]  = ENC_NEG_INF;
}

// ======================= node-tile GEMMs: 5 waves x 16-node tiles ==========
// 30000 nodes = 1875 tiles of 16 = 375 blocks x 5 waves (no tail -> EXEC all 1s)
#define NT_WAVES  5
#define NT_THREADS (NT_WAVES * 32)
#define NT_BLOCKS 375

#if HAVE_WMMA_F32X4
// ---------------- frontend: h = (x0+x1)@W011 + b, full-f32 WMMA ------------
__global__ __launch_bounds__(NT_THREADS) void k_frontend(const float* __restrict__ x,
                                                         const float* __restrict__ W,
                                                         const float* __restrict__ b,
                                                         float* __restrict__ h) {
  __shared__ float sWT[64 * 128];  // W011^T: sWT[n*128+k] = W[k*64+n], 32 KB
  const int tid = threadIdx.x;
  for (int i = tid; i < 64 * 128; i += NT_THREADS) {
    int n = i >> 7, k = i & 127;
    sWT[i] = W[k * 64 + n];
  }
  __syncthreads();

  const int wave = tid >> 5, lane = tid & 31;
  const int m = lane & 15, hi = lane >> 4;
  const int tile  = blockIdx.x * NT_WAVES + wave;   // < 1875
  const int node0 = tile * 16;
  const float* xrow = x + (size_t)(node0 + m) * IN_DIM;  // A row fed by this lane

  v8f acc[4] = {};
  for (int kt = 0; kt < 32; ++kt) {                 // K = 128, 4 per WMMA
    const int kb = kt * 4 + hi * 2;                 // A: lane holds K pair {kb, kb+1}
    v2f A = { xrow[kb] + xrow[128 + kb], xrow[kb + 1] + xrow[128 + kb + 1] };
#pragma unroll
    for (int nt = 0; nt < 4; ++nt) {                // B: column n, K pair {kb, kb+1}
      const float* bp = &sWT[(nt * 16 + m) * 128 + kb];
      v2f B = { bp[0], bp[1] };
      acc[nt] = __builtin_amdgcn_wmma_f32_16x16x4_f32(
          false, A, false, B, (short)0, acc[nt], false, false);
    }
  }
  // D layout: lane holds column n = lane%16; VGPR r holds row M = r + 8*hi
#pragma unroll
  for (int nt = 0; nt < 4; ++nt) {
    const int c = nt * 16 + m;
    const float bias = b[c];
#pragma unroll
    for (int r = 0; r < 8; ++r)
      h[(size_t)(node0 + r + hi * 8) * C0 + c] = acc[nt][r] + bias;
  }
}

// ------- SAGE update: h2 = elu([mean|h] @ [[Wl];[Wr]] + bl), f32 WMMA ------
__global__ __launch_bounds__(NT_THREADS) void k_sage_update(const float* __restrict__ nsum,
                                                            const float* __restrict__ cnt,
                                                            const float* __restrict__ h,
                                                            const float* __restrict__ Wl,
                                                            const float* __restrict__ bl,
                                                            const float* __restrict__ Wr,
                                                            _Float16* __restrict__ h2h) {
  __shared__ float sWT[64 * 128];  // [[Wl];[Wr]]^T: sWT[n*128+k], 32 KB
  const int tid = threadIdx.x;
  for (int i = tid; i < 64 * 128; i += NT_THREADS) {
    int n = i >> 7, k = i & 127;
    sWT[i] = (k < 64) ? Wl[k * 64 + n] : Wr[(k - 64) * 64 + n];
  }
  __syncthreads();

  const int wave = tid >> 5, lane = tid & 31;
  const int m = lane & 15, hi = lane >> 4;
  const int tile  = blockIdx.x * NT_WAVES + wave;
  const int node0 = tile * 16;
  const int node  = node0 + m;
  const float inv = 1.0f / fmaxf(cnt[node], 1.0f);
  const float* nsrow = nsum + (size_t)node * C0;
  const float* hrow  = h    + (size_t)node * C0;

  v8f acc[4] = {};
  for (int kt = 0; kt < 32; ++kt) {
    const int kb = kt * 4 + hi * 2;       // pair never straddles 64; uniform branch
    v2f A;
    if (kb < 64) A = v2f{ nsrow[kb] * inv, nsrow[kb + 1] * inv };
    else         A = v2f{ hrow[kb - 64],   hrow[kb - 63] };
#pragma unroll
    for (int nt = 0; nt < 4; ++nt) {
      const float* bp = &sWT[(nt * 16 + m) * 128 + kb];
      v2f B = { bp[0], bp[1] };
      acc[nt] = __builtin_amdgcn_wmma_f32_16x16x4_f32(
          false, A, false, B, (short)0, acc[nt], false, false);
    }
  }
#pragma unroll
  for (int nt = 0; nt < 4; ++nt) {
    const int c = nt * 16 + m;
    const float bias = bl[c];
#pragma unroll
    for (int r = 0; r < 8; ++r) {
      float v = acc[nt][r] + bias;
      float e = (v > 0.0f) ? v : expm1f(v);  // ELU
      h2h[(size_t)(node0 + r + hi * 8) * C1 + c] = (_Float16)e;
    }
  }
}

#else  // ---------------- VALU fallback paths (no f32 WMMA builtin) ---------
__global__ __launch_bounds__(256) void k_frontend(const float* __restrict__ x,
                                                  const float* __restrict__ W,
                                                  const float* __restrict__ b,
                                                  float* __restrict__ h) {
  __shared__ float sW[128 * 64];
  __shared__ float sx[4][128];
  const int tid = threadIdx.x;
  for (int i = tid; i < 128 * 64; i += 256) sW[i] = W[i];
  const int node0 = blockIdx.x * 4;
  for (int i = tid; i < 4 * 128; i += 256) {
    int r = i >> 7, k = i & 127;
    int n = node0 + r;
    float v = 0.0f;
    if (n < N_NODES) {
      size_t base = (size_t)n * IN_DIM;
      v = x[base + k] + x[base + 128 + k];
    }
    sx[r][k] = v;
  }
  __syncthreads();
  const int c = tid & 63, r = tid >> 6;
  const int n = node0 + r;
  if (n < N_NODES) {
    float acc = b[c];
#pragma unroll 8
    for (int k = 0; k < 128; ++k) acc = fmaf(sx[r][k], sW[k * 64 + c], acc);
    h[(size_t)n * C0 + c] = acc;
  }
}

__global__ __launch_bounds__(256) void k_sage_update(const float* __restrict__ nsum,
                                                     const float* __restrict__ cnt,
                                                     const float* __restrict__ h,
                                                     const float* __restrict__ Wl,
                                                     const float* __restrict__ bl,
                                                     const float* __restrict__ Wr,
                                                     _Float16* __restrict__ h2h) {
  __shared__ float sWl[64 * 64];
  __shared__ float sWr[64 * 64];
  __shared__ float sm[4][64];
  __shared__ float sh[4][64];
  const int tid = threadIdx.x;
  for (int i = tid; i < 64 * 64; i += 256) { sWl[i] = Wl[i]; sWr[i] = Wr[i]; }
  const int node0 = blockIdx.x * 4;
  for (int i = tid; i < 4 * 64; i += 256) {
    int r = i >> 6, k = i & 63;
    int n = node0 + r;
    float mv = 0.0f, hv = 0.0f;
    if (n < N_NODES) {
      float inv = 1.0f / fmaxf(cnt[n], 1.0f);
      mv = nsum[(size_t)n * C0 + k] * inv;
      hv = h[(size_t)n * C0 + k];
    }
    sm[r][k] = mv; sh[r][k] = hv;
  }
  __syncthreads();
  const int c = tid & 63, r = tid >> 6;
  const int n = node0 + r;
  if (n < N_NODES) {
    float acc = bl[c];
#pragma unroll 8
    for (int k = 0; k < 64; ++k)
      acc = fmaf(sm[r][k], sWl[k * 64 + c], fmaf(sh[r][k], sWr[k * 64 + c], acc));
    float e = (acc > 0.0f) ? acc : expm1f(acc);
    h2h[(size_t)n * C1 + c] = (_Float16)e;
  }
}
#endif  // HAVE_WMMA_F32X4

// ---------------------------------- SAGE aggregate: masked scatter-add ------
__global__ __launch_bounds__(256) void k_sage_agg(const int* __restrict__ ei,
                                                  const float* __restrict__ ea,
                                                  const float* __restrict__ h,
                                                  float* __restrict__ nsum,
                                                  float* __restrict__ cnt) {
  int t = blockIdx.x * 256 + threadIdx.x;
  int e = t >> 6;
  int c = t & 63;
  if (e >= N_EDGES) return;
  if (!edge_mask(ea[e])) return;
  int s = ei[e];
  int d = ei[N_EDGES + e];
  atomicAdd(&nsum[(size_t)d * C0 + c], h[(size_t)s * C0 + c]);
  if (c == 0) atomicAdd(&cnt[d], 1.0f);
}

// --------------------------- EdgeConv: WMMA f16 MLP + encoded atomic max ----
__device__ __forceinline__ half8 load_a_chunk(const _Float16* rowi, const _Float16* rowj, int c) {
  // A row = [ xi (64) | xj - xi (64) ], 8-half chunks never straddle 64.
  if (c < C1) return *(const half8*)(rowi + c);
  half8 xj = *(const half8*)(rowj + (c - C1));
  half8 xi = *(const half8*)(rowi + (c - C1));
  return xj - xi;
}

__global__ __launch_bounds__(256) void k_edgeconv(const int* __restrict__ ei,
                                                  const float* __restrict__ ea,
                                                  const _Float16* __restrict__ h2h,
                                                  const float* __restrict__ W1,
                                                  const float* __restrict__ b1,
                                                  const float* __restrict__ W2,
                                                  const float* __restrict__ b2,
                                                  unsigned* __restrict__ agg) {
  __shared__ _Float16 sW1T[64 * 128];   // W1^T (col-major W1), 16 KB
  __shared__ float sW2[128];            // [64][2]
  __shared__ float sB1[64];
  __shared__ float sRelu[8][16 * 65];   // per-wave 16x64 tile, pad 65 -> 33.3 KB

  const int tid = threadIdx.x;
  for (int i = tid; i < 64 * 128; i += 256) {
    int n = i >> 7, k = i & 127;
    sW1T[i] = (_Float16)W1[k * 64 + n];
  }
  if (tid < 128) sW2[tid] = W2[tid];
  if (tid < 64)  sB1[tid] = b1[tid];
  __syncthreads();

  const int wave = tid >> 5;
  const int lane = tid & 31;
  const int m  = lane & 15;   // edge-in-tile / matrix row
  const int hi = lane >> 4;   // lane half selects K sub-range

  const int e = (blockIdx.x * 8 + wave) * 16 + m;   // E divisible by 128
  __builtin_prefetch(&ea[e], 0, 1);                 // global_prefetch_b8: hide late mask load
  const int srcN = ei[e];
  const int dstN = ei[N_EDGES + e];
  const _Float16* rowi = h2h + (size_t)dstN * C1;   // xi = h2[dst]
  const _Float16* rowj = h2h + (size_t)srcN * C1;   // xj = h2[src]

  v8f acc[4] = {};
#pragma unroll
  for (int kt = 0; kt < 4; ++kt) {        // K = 128 in 4 steps of 32
    const int kb = kt * 32 + hi * 8;      // A layout: halves [kb,kb+8) + [kb+16,kb+24)
    half8 alo = load_a_chunk(rowi, rowj, kb);
    half8 ahi = load_a_chunk(rowi, rowj, kb + 16);
    v16h A = __builtin_shufflevector(alo, ahi, 0,1,2,3,4,5,6,7,8,9,10,11,12,13,14,15);
    const int kbB = kt * 32 + hi * 8;     // B layout: column n, halves [kbB,kbB+8) + [kbB+16,kbB+24)
#pragma unroll
    for (int nt = 0; nt < 4; ++nt) {
      const _Float16* bp = &sW1T[(nt * 16 + m) * 128 + kbB];
      half8 blo = *(const half8*)bp;
      half8 bhi = *(const half8*)(bp + 16);
      v16h B = __builtin_shufflevector(blo, bhi, 0,1,2,3,4,5,6,7,8,9,10,11,12,13,14,15);
      acc[nt] = __builtin_amdgcn_wmma_f32_16x16x32_f16(
          /*neg_a=*/false, A, /*neg_b=*/false, B,
          /*c_mod=*/(short)0, acc[nt], /*reuse_a=*/false, /*reuse_b=*/false);
    }
  }

  // bias + ReLU; D layout: lane holds n = m, rows M = r + 8*hi
#pragma unroll
  for (int nt = 0; nt < 4; ++nt) {
    float bias = sB1[nt * 16 + m];
#pragma unroll
    for (int r = 0; r < 8; ++r) {
      float v = fmaxf(acc[nt][r] + bias, 0.0f);
      sRelu[wave][(r + hi * 8) * 65 + nt * 16 + m] = v;
    }
  }
  asm volatile("s_wait_dscnt 0" ::: "memory");  // wave-local LDS visibility

  // tiny second GEMM [16x64]x[64x2] + masked segment-max
  const int j = hi;  // output channel
  float s = b2[j];
  const float* rrow = &sRelu[wave][m * 65];
#pragma unroll
  for (int n = 0; n < 64; ++n) s = fmaf(rrow[n], sW2[n * 2 + j], s);

  if (edge_mask(ea[e]))
    atomicMax(&agg[(size_t)dstN * 2 + j], enc_f32(s));
}

// ------------------------------------------------ finalize: decode + clean --
__global__ __launch_bounds__(256) void k_final(const unsigned* __restrict__ agg,
                                               float* __restrict__ out) {
  int i = blockIdx.x * 256 + threadIdx.x;
  if (i >= N_NODES * 2) return;
  unsigned u = agg[i];
  unsigned fb = (u & 0x80000000u) ? (u & 0x7FFFFFFFu) : ~u;
  out[i] = ((fb & 0x7F800000u) == 0x7F800000u) ? 0.0f : __uint_as_float(fb);
}

// ---------------------------------------------------------------------------
static inline size_t align_up(size_t v) { return (v + 255) & ~(size_t)255; }

extern "C" void kernel_launch(void* const* d_in, const int* in_sizes, int n_in,
                              void* d_out, int out_size, void* d_ws, size_t ws_size,
                              hipStream_t stream) {
  const float* x    = (const float*)d_in[0];
  const int*   ei   = (const int*)  d_in[1];
  const float* ea   = (const float*)d_in[2];
  const float* W011 = (const float*)d_in[3];
  const float* b011 = (const float*)d_in[4];
  const float* Wl   = (const float*)d_in[5];
  const float* bl   = (const float*)d_in[6];
  const float* Wr   = (const float*)d_in[7];
  const float* W1   = (const float*)d_in[8];
  const float* b1   = (const float*)d_in[9];
  const float* W2   = (const float*)d_in[10];
  const float* b2   = (const float*)d_in[11];
  float* out = (float*)d_out;

  char* p = (char*)d_ws;
  float*    h    = (float*)p;    p += align_up((size_t)N_NODES * C0 * 4);
  float*    nsum = (float*)p;    p += align_up((size_t)N_NODES * C0 * 4);
  float*    cnt  = (float*)p;    p += align_up((size_t)N_NODES * 4);
  _Float16* h2h  = (_Float16*)p; p += align_up((size_t)N_NODES * C1 * 2);
  unsigned* agg  = (unsigned*)p; p += align_up((size_t)N_NODES * 2 * 4);

  k_init<<<(N_NODES * C0 + 255) / 256, 256, 0, stream>>>(nsum, cnt, agg);
#if HAVE_WMMA_F32X4
  k_frontend   <<<NT_BLOCKS, NT_THREADS, 0, stream>>>(x, W011, b011, h);
#else
  k_frontend   <<<(N_NODES + 3) / 4, 256, 0, stream>>>(x, W011, b011, h);
#endif
  k_sage_agg   <<<(N_EDGES * 64) / 256, 256, 0, stream>>>(ei, ea, h, nsum, cnt);
#if HAVE_WMMA_F32X4
  k_sage_update<<<NT_BLOCKS, NT_THREADS, 0, stream>>>(nsum, cnt, h, Wl, bl, Wr, h2h);
#else
  k_sage_update<<<(N_NODES + 3) / 4, 256, 0, stream>>>(nsum, cnt, h, Wl, bl, Wr, h2h);
#endif
  k_edgeconv   <<<N_EDGES / 128, 256, 0, stream>>>(ei, ea, h2h, W1, b1, W2, b2, agg);
  k_final      <<<(N_NODES * 2 + 255) / 256, 256, 0, stream>>>(agg, out);
}
